// LruLM_11501922418995
// MI455X (gfx1250) — matrix operand order repaired and checked by reading, not compiled, not run on hardware
//
#include <hip/hip_runtime.h>
#include <hip/hip_bf16.h>
#include <math.h>

// ---------------- model constants ----------------
#define B_ 2
#define S_ 2048
#define D_ 768
#define N_ (B_ * S_)   // 4096 rows
#define H_ 3072
#define V_ 50257
#define VP_ 50304       // V padded to multiple of 128
#define L_ 6
#define NC_ 64          // scan chunks
#define CL_ (S_ / NC_)  // 32 steps per chunk

typedef __attribute__((ext_vector_type(16))) __bf16          v16bf;
typedef __attribute__((ext_vector_type(16))) unsigned short  v16u;
typedef __attribute__((ext_vector_type(8)))  float           v8f;

// fp32 -> bf16 round-to-nearest-even
__device__ __forceinline__ unsigned short f2bf(float f) {
    unsigned u = __builtin_bit_cast(unsigned, f);
    unsigned r = u + 0x7FFFu + ((u >> 16) & 1u);
    return (unsigned short)(r >> 16);
}

// 16-byte global -> LDS async copy (CDNA5 async engine, tracked by ASYNCcnt).
// LDS byte offset = low 32 bits of the generic pointer (ISA 10.2 aperture rule).
__device__ __forceinline__ void async_cp16(void* lds_ptr, const void* gptr) {
    unsigned lds = (unsigned)(unsigned long long)lds_ptr;
    asm volatile("global_load_async_to_lds_b128 %0, %1, off"
                 :: "v"(lds), "v"(gptr) : "memory");
}
__device__ __forceinline__ void wait_async0() {
    asm volatile("s_wait_asynccnt 0x0" ::: "memory");
}

// ---------------- embedding gather ----------------
__global__ __launch_bounds__(256) void embed_kernel(
    const int* __restrict__ tokens, const float* __restrict__ emb,
    float* __restrict__ x)
{
    int i = blockIdx.x * 256 + threadIdx.x;
    if (i >= N_ * D_) return;
    int n = i / D_, d = i % D_;
    x[i] = emb[(size_t)tokens[n] * D_ + d];
}

// ---------------- weight convert + transpose: fp32 (K x N) -> bf16 (Np x K) --
// Np is N padded to a multiple of 128; pad rows are written as zeros so the
// GEMM can load B tiles unguarded.
__global__ __launch_bounds__(256) void convtr_kernel(
    const float* __restrict__ in, unsigned short* __restrict__ outp,
    int K, int Nn)
{
    __shared__ unsigned short t[32][33];
    int n0 = blockIdx.x * 32;
    int k0 = blockIdx.y * 32;
    int lx = threadIdx.x & 31, ly = threadIdx.x >> 5;
    #pragma unroll
    for (int r = ly; r < 32; r += 8) {
        int k = k0 + r, n = n0 + lx;
        float f = (n < Nn) ? in[(size_t)k * Nn + n] : 0.f;  // K multiple of 32
        t[r][lx] = f2bf(f);
    }
    __syncthreads();
    #pragma unroll
    for (int r = ly; r < 32; r += 8) {
        int n = n0 + r, k = k0 + lx;
        outp[(size_t)n * K + k] = t[lx][r];
    }
}

// ---------------- layernorm (fp32 in, bf16 out) ----------------
template <int C>
__global__ __launch_bounds__(256) void ln_kernel(
    const float* __restrict__ x, const float* __restrict__ g,
    const float* __restrict__ b, unsigned short* __restrict__ out, int ldo)
{
    int row = blockIdx.x;
    const float* xr = x + (size_t)row * C;
    float s = 0.f, s2 = 0.f;
    for (int i = threadIdx.x; i < C; i += 256) { float v = xr[i]; s += v; s2 += v * v; }
    __shared__ float sh0[256], sh1[256];
    sh0[threadIdx.x] = s; sh1[threadIdx.x] = s2;
    __syncthreads();
    for (int off = 128; off > 0; off >>= 1) {
        if (threadIdx.x < off) {
            sh0[threadIdx.x] += sh0[threadIdx.x + off];
            sh1[threadIdx.x] += sh1[threadIdx.x + off];
        }
        __syncthreads();
    }
    float m   = sh0[0] / C;
    float var = sh1[0] / C - m * m;
    float inv = rsqrtf(var + 1e-5f);
    for (int i = threadIdx.x; i < C; i += 256) {
        float v = (xr[i] - m) * inv * g[i] + b[i];
        out[(size_t)row * ldo + i] = f2bf(v);
    }
}

// ---------------- WMMA bf16 GEMM ----------------
// A:  bf16 (M x K, row-major, lda)
// Bt: bf16 (Npad x K, row-major, ld = K)  — weights pre-transposed, pad rows 0
// Out = epi(A@B + bias) [+res].  128x128x32 tile, 8 wave32 waves, each wave a
// 32x64 patch = 2x4 v_wmma_f32_16x16x32_bf16. Double-buffered LDS tiles filled
// by global_load_async_to_lds_b128, synced with s_wait_asynccnt.
template <int EPI /*0=none,1=gelu*/, bool RES, bool OBF16>
__global__ __launch_bounds__(256) void gemm_bf16_kernel(
    const unsigned short* __restrict__ A, int lda,
    const unsigned short* __restrict__ Bt,
    const float* __restrict__ bias,
    const float* __restrict__ res, int ldres,
    void* __restrict__ Outv, int ldo,
    int M, int Nn, int K)
{
    __shared__ unsigned short As[2][128][40]; // [buf][row][k], +8 pad (80B rows)
    __shared__ unsigned short Bs[2][128][40]; // [buf][col][k]

    const int n0   = blockIdx.x * 128;
    const int m0   = blockIdx.y * 128;
    const int tid  = threadIdx.x;
    const int lane = tid & 31;
    const int wave = tid >> 5;
    const int wm   = wave >> 1;  // 0..3 : row quadrant (32 rows)
    const int wn   = wave & 1;   // 0..1 : col half (64 cols)

    // stage one 128x32 bf16 tile (512 x 16B segments, 2 per thread)
    auto stage = [&](unsigned short (*dst)[40], const unsigned short* src,
                     int ld, int r0, int k0) {
        #pragma unroll
        for (int s = tid; s < 512; s += 256) {
            int row = s >> 2, cb = (s & 3) * 8;
            async_cp16(&dst[row][cb], src + (size_t)(r0 + row) * ld + k0 + cb);
        }
    };

    v8f acc[2][4] = {};

    stage(As[0], A, lda, m0, 0);
    stage(Bs[0], Bt, K, n0, 0);
    wait_async0();
    __syncthreads();

    int cur = 0;
    for (int k0 = 0; k0 < K; k0 += 32) {
        int nxt = cur ^ 1;
        if (k0 + 32 < K) {                    // prefetch next tile via async DMA
            stage(As[nxt], A, lda, m0, k0 + 32);
            stage(Bs[nxt], Bt, K, n0, k0 + 32);
        }

        // A fragments: lanes 0-15 = rows, lane>=16 selects K-half (+8);
        // elements 0..7 -> K=kh..kh+7, elements 8..15 -> K=16+kh..
        v16u afr[2], bfr[4];
        const int kh = (lane >> 4) * 8;
        #pragma unroll
        for (int r = 0; r < 2; r++) {
            int row = wm * 32 + r * 16 + (lane & 15);
            #pragma unroll
            for (int i = 0; i < 8; i++) afr[r][i] = As[cur][row][kh + i];
            #pragma unroll
            for (int i = 0; i < 8; i++) afr[r][8 + i] = As[cur][row][16 + kh + i];
        }
        // B fragments: col = lane&15, same K order as A (lane>=16 selects +8).
        const int kb = (lane >> 4) * 8;
        #pragma unroll
        for (int c = 0; c < 4; c++) {
            int col = wn * 64 + c * 16 + (lane & 15);
            #pragma unroll
            for (int i = 0; i < 8; i++) bfr[c][i] = Bs[cur][col][kb + i];
            #pragma unroll
            for (int i = 0; i < 8; i++) bfr[c][8 + i] = Bs[cur][col][16 + kb + i];
        }
        #pragma unroll
        for (int r = 0; r < 2; r++)
            #pragma unroll
            for (int c = 0; c < 4; c++)
                acc[r][c] = __builtin_amdgcn_wmma_f32_16x16x32_bf16(
                    false, __builtin_bit_cast(v16bf, afr[r]),
                    false, __builtin_bit_cast(v16bf, bfr[c]),
                    (short)0, acc[r][c], false, false);

        wait_async0();        // prefetch landed (ran concurrently with WMMAs)
        __syncthreads();
        cur = nxt;
    }

    // epilogue: C/D layout — VGPR v holds M = v + 8*(lane>>4), N = lane&15
    #pragma unroll
    for (int r = 0; r < 2; r++) {
        int rbase = m0 + wm * 32 + r * 16 + (lane >> 4) * 8;
        #pragma unroll
        for (int c = 0; c < 4; c++) {
            int colg = n0 + wn * 64 + c * 16 + (lane & 15);
            if (colg < Nn) {
                float bv = bias[colg];
                #pragma unroll
                for (int v = 0; v < 8; v++) {
                    int rowg = rbase + v;
                    float val = acc[r][c][v] + bv;
                    if (EPI == 1)
                        val = 0.5f * val * (1.0f + erff(val * 0.70710678118654752f));
                    if (RES) val += res[(size_t)rowg * ldres + colg];
                    if (OBF16)
                        ((unsigned short*)Outv)[(size_t)rowg * ldo + colg] = f2bf(val);
                    else
                        ((float*)Outv)[(size_t)rowg * ldo + colg] = val;
                }
            }
        }
    }
}

// ---------------- LRU complex scan, 3-pass chunked ----------------
__device__ __forceinline__ void lru_f(const float* nu_log, const float* th_log,
                                      int d, float& fr, float& fi)
{
    float nu = expf(-expf(nu_log[d]));
    float th = expf(th_log[d]);
    fr = nu * cosf(th);
    fi = nu * sinf(th);
}

__global__ __launch_bounds__(256) void scan1_kernel(
    const float* __restrict__ u, const float* __restrict__ nu_log,
    const float* __restrict__ th_log, const float* __restrict__ ga_log,
    float* __restrict__ y, float* __restrict__ carr)
{
    int tid = blockIdx.x * 256 + threadIdx.x;
    if (tid >= B_ * NC_ * D_) return;
    int d = tid % D_;
    int c = (tid / D_) % NC_;
    int b = tid / (D_ * NC_);
    float fr, fi; lru_f(nu_log, th_log, d, fr, fi);
    float ga = expf(ga_log[d]);
    float yr = 0.f, yi = 0.f;
    int t0 = c * CL_;
    const float* up = u + (size_t)(b * S_ + t0) * (4 * D_);
    float*       yp = y + (size_t)(b * S_ + t0) * (2 * D_);
    for (int k = 0; k < CL_; k++) {
        float xr = ga * up[d];
        float xi = ga * up[D_ + d];
        float nyr = fr * yr - fi * yi + xr;
        float nyi = fr * yi + fi * yr + xi;
        yr = nyr; yi = nyi;
        yp[d] = yr; yp[D_ + d] = yi;
        up += 4 * D_; yp += 2 * D_;
    }
    int idx = ((b * D_ + d) * NC_ + c) * 2;
    carr[idx] = yr; carr[idx + 1] = yi;
}

__global__ __launch_bounds__(256) void scan2_kernel(
    const float* __restrict__ carr, const float* __restrict__ nu_log,
    const float* __restrict__ th_log, float* __restrict__ carrPre)
{
    int tid = blockIdx.x * 256 + threadIdx.x;
    if (tid >= B_ * D_) return;
    int d = tid % D_, b = tid / D_;
    float fr, fi; lru_f(nu_log, th_log, d, fr, fi);
    float pr = 1.f, pi = 0.f;               // f^CL_
    for (int i = 0; i < CL_; i++) {
        float nr = pr * fr - pi * fi;
        pi = pr * fi + pi * fr; pr = nr;
    }
    float cr = 0.f, ci = 0.f;
    for (int c = 0; c < NC_; c++) {
        int idx = ((b * D_ + d) * NC_ + c) * 2;
        carrPre[idx] = cr; carrPre[idx + 1] = ci;   // carry entering chunk c
        float lr = carr[idx], li = carr[idx + 1];
        float nr = pr * cr - pi * ci + lr;
        float ni = pr * ci + pi * cr + li;
        cr = nr; ci = ni;
    }
}

__global__ __launch_bounds__(256) void scan3_kernel(
    float* __restrict__ y, const float* __restrict__ nu_log,
    const float* __restrict__ th_log, const float* __restrict__ carrPre)
{
    int tid = blockIdx.x * 256 + threadIdx.x;
    if (tid >= B_ * NC_ * D_) return;
    int d = tid % D_;
    int c = (tid / D_) % NC_;
    int b = tid / (D_ * NC_);
    float fr, fi; lru_f(nu_log, th_log, d, fr, fi);
    int idx = ((b * D_ + d) * NC_ + c) * 2;
    float cr = carrPre[idx], ci = carrPre[idx + 1];
    int t0 = c * CL_;
    float* yp = y + (size_t)(b * S_ + t0) * (2 * D_);
    for (int k = 0; k < CL_; k++) {
        float nr = fr * cr - fi * ci;
        ci = fr * ci + fi * cr; cr = nr;
        yp[d]      += cr;
        yp[D_ + d] += ci;
        yp += 2 * D_;
    }
}

__global__ __launch_bounds__(256) void gate_kernel(
    float* __restrict__ y, const float* __restrict__ u)
{
    long long i = (long long)blockIdx.x * 256 + threadIdx.x;
    if (i >= (long long)N_ * (2 * D_)) return;
    int n = (int)(i / (2 * D_));
    int j = (int)(i % (2 * D_));
    float o = u[(size_t)n * (4 * D_) + 2 * D_ + j];
    float s = o / (1.f + expf(-o));
    y[i] *= s;
}

// ---------------- host driver ----------------
extern "C" void kernel_launch(void* const* d_in, const int* in_sizes, int n_in,
                              void* d_out, int out_size, void* d_ws, size_t ws_size,
                              hipStream_t stream) {
    const int*   tokens    = (const int*)  d_in[0];
    const float* emb       = (const float*)d_in[1];
    const float* ln1_g     = (const float*)d_in[2];
    const float* ln1_b     = (const float*)d_in[3];
    const float* inw       = (const float*)d_in[4];
    const float* inb       = (const float*)d_in[5];
    const float* nu_log    = (const float*)d_in[6];
    const float* theta_log = (const float*)d_in[7];
    const float* gamma_log = (const float*)d_in[8];
    const float* lnr_g     = (const float*)d_in[9];
    const float* lnr_b     = (const float*)d_in[10];
    const float* outw      = (const float*)d_in[11];
    const float* outb      = (const float*)d_in[12];
    const float* ln2_g     = (const float*)d_in[13];
    const float* ln2_b     = (const float*)d_in[14];
    const float* w1        = (const float*)d_in[15];
    const float* b1        = (const float*)d_in[16];
    const float* w2        = (const float*)d_in[17];
    const float* b2        = (const float*)d_in[18];
    const float* lnf_g     = (const float*)d_in[19];
    const float* lnf_b     = (const float*)d_in[20];
    const float* pw        = (const float*)d_in[21];
    const float* pb        = (const float*)d_in[22];
    float* out = (float*)d_out;

    char* w = (char*)d_ws;
    auto alloc = [&](size_t bytes) -> void* {
        void* p = (void*)w;
        w += (bytes + 255) & ~(size_t)255;
        return p;
    };
    float*          x    = (float*)alloc((size_t)N_ * D_ * 4);            // residual
    unsigned short* abf  = (unsigned short*)alloc((size_t)N_ * 1536 * 2); // bf16 LN out
    float*          u    = (float*)alloc((size_t)N_ * 3072 * 4);          // in-proj out
    float*          y    = (float*)alloc((size_t)N_ * 1536 * 4);          // scan out
    unsigned short* hbf  = (unsigned short*)alloc((size_t)N_ * 3072 * 2); // MLP hidden
    unsigned short* wbf  = (unsigned short*)alloc((size_t)VP_ * D_ * 2);  // bf16 weight (transposed)
    float*          carr = (float*)alloc((size_t)B_ * D_ * NC_ * 2 * 4);
    float*          cpre = (float*)alloc((size_t)B_ * D_ * NC_ * 2 * 4);

    const int MT  = N_ / 128;            // 32 row tiles
    const int nsc = B_ * NC_ * D_;       // scan threads

    embed_kernel<<<(N_ * D_ + 255) / 256, 256, 0, stream>>>(tokens, emb, x);

    for (int i = 0; i < L_; i++) {
        // ---- LRU branch ----
        ln_kernel<D_><<<N_, 256, 0, stream>>>(x, ln1_g + i * D_, ln1_b + i * D_, abf, D_);
        convtr_kernel<<<dim3(3072 / 32, D_ / 32), 256, 0, stream>>>(
            inw + (size_t)i * D_ * 3072, wbf, D_, 3072);
        gemm_bf16_kernel<0, false, false><<<dim3(3072 / 128, MT), 256, 0, stream>>>(
            abf, D_, wbf, inb + i * 3072, nullptr, 0, u, 3072, N_, 3072, D_);
        scan1_kernel<<<(nsc + 255) / 256, 256, 0, stream>>>(
            u, nu_log + i * D_, theta_log + i * D_, gamma_log + i * D_, y, carr);
        scan2_kernel<<<(B_ * D_ + 255) / 256, 256, 0, stream>>>(
            carr, nu_log + i * D_, theta_log + i * D_, cpre);
        scan3_kernel<<<(nsc + 255) / 256, 256, 0, stream>>>(
            y, nu_log + i * D_, theta_log + i * D_, cpre);
        gate_kernel<<<(int)(((long long)N_ * 1536 + 255) / 256), 256, 0, stream>>>(y, u);
        ln_kernel<1536><<<N_, 256, 0, stream>>>(y, lnr_g + i * 1536, lnr_b + i * 1536, abf, 1536);
        convtr_kernel<<<dim3(D_ / 32, 1536 / 32), 256, 0, stream>>>(
            outw + (size_t)i * 1536 * D_, wbf, 1536, D_);
        gemm_bf16_kernel<0, true, false><<<dim3(D_ / 128, MT), 256, 0, stream>>>(
            abf, 1536, wbf, outb + i * D_, x, D_, x, D_, N_, D_, 1536);
        // ---- MLP branch ----
        ln_kernel<D_><<<N_, 256, 0, stream>>>(x, ln2_g + i * D_, ln2_b + i * D_, abf, D_);
        convtr_kernel<<<dim3(H_ / 32, D_ / 32), 256, 0, stream>>>(
            w1 + (size_t)i * D_ * H_, wbf, D_, H_);
        gemm_bf16_kernel<1, false, true><<<dim3(H_ / 128, MT), 256, 0, stream>>>(
            abf, D_, wbf, b1 + i * H_, nullptr, 0, hbf, H_, N_, H_, D_);
        convtr_kernel<<<dim3(D_ / 32, H_ / 32), 256, 0, stream>>>(
            w2 + (size_t)i * H_ * D_, wbf, H_, D_);
        gemm_bf16_kernel<0, true, false><<<dim3(D_ / 128, MT), 256, 0, stream>>>(
            hbf, H_, wbf, b2 + i * D_, x, D_, x, D_, N_, D_, H_);
    }

    // ---- final LN + vocab projection ----
    ln_kernel<D_><<<N_, 256, 0, stream>>>(x, lnf_g, lnf_b, abf, D_);
    convtr_kernel<<<dim3(VP_ / 32, D_ / 32), 256, 0, stream>>>(pw, wbf, D_, V_);
    gemm_bf16_kernel<0, false, false><<<dim3(VP_ / 128, MT), 256, 0, stream>>>(
        abf, D_, wbf, pb, nullptr, 0, out, V_, N_, V_, D_);
}
